// Cross_MultiAttention_11244224381182
// MI455X (gfx1250) — compile-verified, matrix operand order checked by reading, not run on hardware
//
#include <hip/hip_runtime.h>
#include <hip/hip_bf16.h>
#include <math.h>

// ---------------------------------------------------------------------------
// Cross multi-head attention for MI455X (gfx1250, wave32, WMMA bf16/f32-acc).
//   pipeline: fold proj_in into q/k/v weights -> token prep (x+pos, bf16,
//   transposed) -> Q/K GEMM, V^T GEMM -> flash attention (8 heads = 8 waves
//   per workgroup, mask read once per query tile) -> output GEMM.
// All matrix math through __builtin_amdgcn_wmma_f32_16x16x32_bf16.
// ---------------------------------------------------------------------------

#define EMB   256
#define HEADS 8
#define DEPTH 32
#define NTOK  5000
#define NPAD  5024      // multiple of 32 (157 key chunks)
#define HH    50
#define WW    100

typedef __attribute__((ext_vector_type(16))) __bf16 bf16x16;
typedef __attribute__((ext_vector_type(8)))  float  f32x8;

struct U128x2 { uint4 a, b; };

__device__ __forceinline__ unsigned short f2bf(float f) {
  unsigned int x = __builtin_bit_cast(unsigned int, f);
  x += 0x7fffu + ((x >> 16) & 1u);          // round-to-nearest-even
  return (unsigned short)(x >> 16);
}
__device__ __forceinline__ unsigned int pack2(float a, float b) {
  return (unsigned int)f2bf(a) | ((unsigned int)f2bf(b) << 16);
}
// Load a WMMA 16-bit operand: two contiguous 16B chunks (per-lane layout).
__device__ __forceinline__ bf16x16 ld16(const unsigned short* p0,
                                        const unsigned short* p1) {
  U128x2 u;
  u.a = *reinterpret_cast<const uint4*>(p0);
  u.b = *reinterpret_cast<const uint4*>(p1);
  return __builtin_bit_cast(bf16x16, u);
}
__device__ __forceinline__ f32x8 wmma_bf16(bf16x16 a, bf16x16 b, f32x8 c) {
  return __builtin_amdgcn_wmma_f32_16x16x32_bf16(false, a, false, b,
                                                 (short)0, c, false, false);
}

// ---------------------------------------------------------------------------
// Fold proj_in into q/k/v: Wq' = wq @ w_in (bf16, [e][c]), bq' = wq@b_in + bq.
// Also bf16-convert w_out.
// ---------------------------------------------------------------------------
__global__ __launch_bounds__(256) void prep_weights(
    const float* __restrict__ wq, const float* __restrict__ bq,
    const float* __restrict__ wk, const float* __restrict__ bk,
    const float* __restrict__ wv, const float* __restrict__ bv,
    const float* __restrict__ w_in, const float* __restrict__ b_in,
    const float* __restrict__ w_out,
    unsigned short* Wq, unsigned short* Wk, unsigned short* Wv,
    unsigned short* Wo, float* bq2, float* bk2, float* bv2) {
  int e = blockIdx.x, c = threadIdx.x;
  float aq = 0.f, ak = 0.f, av = 0.f;
  for (int k = 0; k < EMB; ++k) {
    float wi = w_in[k * EMB + c];
    aq += wq[e * EMB + k] * wi;
    ak += wk[e * EMB + k] * wi;
    av += wv[e * EMB + k] * wi;
  }
  Wq[e * EMB + c] = f2bf(aq);
  Wk[e * EMB + c] = f2bf(ak);
  Wv[e * EMB + c] = f2bf(av);
  Wo[e * EMB + c] = f2bf(w_out[e * EMB + c]);
  if (c == 0) { float s = bq[e]; for (int k = 0; k < EMB; ++k) s += wq[e*EMB+k]*b_in[k]; bq2[e] = s; }
  if (c == 1) { float s = bk[e]; for (int k = 0; k < EMB; ++k) s += wk[e*EMB+k]*b_in[k]; bk2[e] = s; }
  if (c == 2) { float s = bv[e]; for (int k = 0; k < EMB; ++k) s += wv[e*EMB+k]*b_in[k]; bv2[e] = s; }
}

// ---------------------------------------------------------------------------
// Token prep: A[n][c] = bf16(img[c][n] + pos[c][n]), transposed via LDS tile.
// Rows n in [NTOK, NPAD) are written as zero.
// ---------------------------------------------------------------------------
__global__ __launch_bounds__(256) void prep_tokens(
    const float* __restrict__ x, const float* __restrict__ ctx,
    const float* __restrict__ row_embed, const float* __restrict__ col_embed,
    unsigned short* A_x, unsigned short* A_c) {
  __shared__ float tile[32][33];
  const float* src = blockIdx.z ? ctx : x;
  unsigned short* dst = blockIdx.z ? A_c : A_x;
  int nb = blockIdx.x * 32, cb = blockIdx.y * 32;
  int tn = threadIdx.x & 31, th = threadIdx.x >> 5;   // 0..31, 0..7
  for (int i = 0; i < 4; ++i) {
    int cl = th + i * 8, c = cb + cl, n = nb + tn;
    float v = 0.f;
    if (n < NTOK) {
      float pv = (c < 128) ? col_embed[(n % WW) * 128 + c]
                           : row_embed[(n / WW) * 128 + (c - 128)];
      v = src[(size_t)c * NTOK + n] + pv;
    }
    tile[cl][tn] = v;
  }
  __syncthreads();
  for (int i = 0; i < 4; ++i) {
    int nl = th + i * 8;
    dst[(size_t)(nb + nl) * EMB + cb + tn] = f2bf(tile[tn][nl]);
  }
}

// ---------------------------------------------------------------------------
// D[n][e] = A[n][c] x W[e][c]^T + bias, times scale.   (Q and K)
// 8 waves: wave w -> e-tiles {2w,2w+1}. grid.x = NPAD/16.
// ---------------------------------------------------------------------------
__global__ __launch_bounds__(256) void gemm_ne(
    const unsigned short* __restrict__ A, const unsigned short* __restrict__ W,
    const float* __restrict__ bias, float scale, unsigned short* dst) {
  int nb = blockIdx.x * 16;
  int wave = threadIdx.x >> 5, lane = threadIdx.x & 31;
  int l16 = lane & 15, lh = lane >> 4;
  int eb0 = wave * 32, eb1 = eb0 + 16;
  f32x8 acc0 = {0.f,0.f,0.f,0.f,0.f,0.f,0.f,0.f};
  f32x8 acc1 = acc0;
  size_t arow = (size_t)(nb + l16) * EMB;
  size_t b0r = (size_t)(eb0 + l16) * EMB, b1r = (size_t)(eb1 + l16) * EMB;
  for (int kc = 0; kc < 8; ++kc) {
    int c0 = kc * 32 + 8 * lh;
    bf16x16 a  = ld16(A + arow + c0, A + arow + c0 + 16);
    bf16x16 b0 = ld16(W + b0r + c0, W + b0r + c0 + 16);
    bf16x16 b1 = ld16(W + b1r + c0, W + b1r + c0 + 16);
    acc0 = wmma_bf16(a, b0, acc0);
    acc1 = wmma_bf16(a, b1, acc1);
  }
  float bv0 = bias[eb0 + l16], bv1 = bias[eb1 + l16];
#pragma unroll
  for (int r = 0; r < 8; ++r) {
    size_t row = (size_t)(nb + r + 8 * lh) * EMB;
    dst[row + eb0 + l16] = f2bf((acc0[r] + bv0) * scale);
    dst[row + eb1 + l16] = f2bf((acc1[r] + bv1) * scale);
  }
}

// ---------------------------------------------------------------------------
// D[e][n] = Wt[e][c] x A[n][c]^T + bias[e].  Used for V^T (bf16 out).
// ---------------------------------------------------------------------------
__global__ __launch_bounds__(256) void gemm_vt(
    const unsigned short* __restrict__ Wt, const unsigned short* __restrict__ A,
    const float* __restrict__ bias, unsigned short* dst) {
  int nb = blockIdx.x * 16;
  int wave = threadIdx.x >> 5, lane = threadIdx.x & 31;
  int l16 = lane & 15, lh = lane >> 4;
  int eb0 = wave * 32, eb1 = eb0 + 16;
  f32x8 acc0 = {0.f,0.f,0.f,0.f,0.f,0.f,0.f,0.f};
  f32x8 acc1 = acc0;
  size_t brow = (size_t)(nb + l16) * EMB;
  size_t a0r = (size_t)(eb0 + l16) * EMB, a1r = (size_t)(eb1 + l16) * EMB;
  for (int kc = 0; kc < 8; ++kc) {
    int c0 = kc * 32 + 8 * lh;
    bf16x16 b  = ld16(A + brow + c0, A + brow + c0 + 16);
    bf16x16 a0 = ld16(Wt + a0r + c0, Wt + a0r + c0 + 16);
    bf16x16 a1 = ld16(Wt + a1r + c0, Wt + a1r + c0 + 16);
    acc0 = wmma_bf16(a0, b, acc0);
    acc1 = wmma_bf16(a1, b, acc1);
  }
#pragma unroll
  for (int r = 0; r < 8; ++r) {
    int e0 = eb0 + r + 8 * lh, e1 = eb1 + r + 8 * lh;
    dst[(size_t)e0 * NPAD + nb + l16] = f2bf(acc0[r] + bias[e0]);
    dst[(size_t)e1 * NPAD + nb + l16] = f2bf(acc1[r] + bias[e1]);
  }
}

// ---------------------------------------------------------------------------
// out[c][n] = Wo[c][e] x O[n][e]^T + b_out[c]   (f32 output, n < NTOK)
// ---------------------------------------------------------------------------
__global__ __launch_bounds__(256) void gemm_out(
    const unsigned short* __restrict__ Wo, const unsigned short* __restrict__ O,
    const float* __restrict__ b_out, float* __restrict__ out) {
  int nb = blockIdx.x * 16;
  int wave = threadIdx.x >> 5, lane = threadIdx.x & 31;
  int l16 = lane & 15, lh = lane >> 4;
  int eb0 = wave * 32, eb1 = eb0 + 16;
  f32x8 acc0 = {0.f,0.f,0.f,0.f,0.f,0.f,0.f,0.f};
  f32x8 acc1 = acc0;
  size_t brow = (size_t)(nb + l16) * EMB;
  size_t a0r = (size_t)(eb0 + l16) * EMB, a1r = (size_t)(eb1 + l16) * EMB;
  for (int kc = 0; kc < 8; ++kc) {
    int c0 = kc * 32 + 8 * lh;
    bf16x16 b  = ld16(O + brow + c0, O + brow + c0 + 16);
    bf16x16 a0 = ld16(Wo + a0r + c0, Wo + a0r + c0 + 16);
    bf16x16 a1 = ld16(Wo + a1r + c0, Wo + a1r + c0 + 16);
    acc0 = wmma_bf16(a0, b, acc0);
    acc1 = wmma_bf16(a1, b, acc1);
  }
  int n = nb + l16;
  if (n < NTOK) {
#pragma unroll
    for (int r = 0; r < 8; ++r) {
      int e0 = eb0 + r + 8 * lh, e1 = eb1 + r + 8 * lh;
      out[(size_t)e0 * NTOK + n] = acc0[r] + b_out[e0];
      out[(size_t)e1 * NTOK + n] = acc1[r] + b_out[e1];
    }
  }
}

// ---------------------------------------------------------------------------
// Flash attention. One workgroup = one 16-query tile, 8 waves = 8 heads
// (mask tile fetched once into WGP$, shared by all heads).
// Computes S^T = K . Q^T so the key-axis softmax reduction is in-lane + one
// shfl_xor(16), and exp(S^T) maps lane-locally onto the PV B-operand.
// Q is pre-scaled by EMB^-0.5. Keys >= NTOK forced to -1e30.
// ---------------------------------------------------------------------------
__global__ __launch_bounds__(256) void flash_attn(
    const unsigned short* __restrict__ Q, const unsigned short* __restrict__ K,
    const unsigned short* __restrict__ Vt,
    const unsigned char* __restrict__ mask, unsigned short* __restrict__ O) {
  int qb = blockIdx.x * 16;
  int h = threadIdx.x >> 5;                 // wave = head
  int lane = threadIdx.x & 31, l16 = lane & 15, lh = lane >> 4;
  int hb = h * DEPTH;
  int q = qb + l16;

  const unsigned short* qrow = Q + (size_t)q * EMB + hb;
  bf16x16 Bq = ld16(qrow + 8 * lh, qrow + 16 + 8 * lh);   // Q^T B operand

  f32x8 acc0 = {0.f,0.f,0.f,0.f,0.f,0.f,0.f,0.f};        // O^T rows d=0..15
  f32x8 acc1 = acc0;                                      // O^T rows d=16..31
  float m = -1e30f, l = 0.f;
  int qm = (q < NTOK) ? q : (NTOK - 1);
  const unsigned char* mrow = mask + (size_t)qm * NTOK;

  const unsigned short* v0base = Vt + (size_t)(hb + l16) * NPAD;
  const unsigned short* v1base = Vt + (size_t)(hb + 16 + l16) * NPAD;

  for (int jb = 0; jb < NPAD; jb += 32) {
    __builtin_prefetch(mrow + ((jb + 32 < NTOK) ? jb + 32 : 0), 0, 1);
    f32x8 S[2];
#pragma unroll
    for (int s = 0; s < 2; ++s) {
      int key = jb + s * 16 + l16;
      const unsigned short* krow = K + (size_t)key * EMB + hb;
      bf16x16 Ak = ld16(krow + 8 * lh, krow + 16 + 8 * lh);
      f32x8 z = {0.f,0.f,0.f,0.f,0.f,0.f,0.f,0.f};
      S[s] = wmma_bf16(Ak, Bq, z);          // S^T tile: rows=keys, cols=queries
    }
    // mask + running max (keys for this lane: kb8..kb8+7 per sub-chunk)
    float cmax = -1e30f;
#pragma unroll
    for (int s = 0; s < 2; ++s) {
      int kb8 = jb + s * 16 + 8 * lh;
      int kc = (kb8 > NTOK - 8) ? (NTOK - 8) : kb8;       // 8B-aligned, safe
      unsigned long long mb = *reinterpret_cast<const unsigned long long*>(mrow + kc);
#pragma unroll
      for (int r = 0; r < 8; ++r) {
        int key = kb8 + r;
        float v = S[s][r];
        if ((mb >> (8 * r)) & 0xffull) v = -1e9f;         // pad_mask -> -1e9
        if (key >= NTOK) v = -1e30f;                      // padded keys
        S[s][r] = v;
        cmax = fmaxf(cmax, v);
      }
    }
    cmax = fmaxf(cmax, __shfl_xor(cmax, 16, 32));
    float mn = fmaxf(m, cmax);
    float alpha = __expf(m - mn);
    m = mn;
#pragma unroll
    for (int r = 0; r < 8; ++r) { acc0[r] *= alpha; acc1[r] *= alpha; }
    l *= alpha;

    bf16x16 Pb;                             // P^T B operand (lane-local pack)
    float ps = 0.f;
#pragma unroll
    for (int s = 0; s < 2; ++s)
#pragma unroll
      for (int r = 0; r < 8; ++r) {
        float p = __expf(S[s][r] - mn);
        ps += p;
        Pb[s * 8 + r] = __builtin_bit_cast(__bf16, f2bf(p));
      }
    ps += __shfl_xor(ps, 16, 32);
    l += ps;

    bf16x16 Av0 = ld16(v0base + jb + 8 * lh, v0base + jb + 16 + 8 * lh);
    bf16x16 Av1 = ld16(v1base + jb + 8 * lh, v1base + jb + 16 + 8 * lh);
    acc0 = wmma_bf16(Av0, Pb, acc0);
    acc1 = wmma_bf16(Av1, Pb, acc1);
  }

  float rl = 1.f / l;
  uint4 o0, o1;
  o0.x = pack2(acc0[0]*rl, acc0[1]*rl); o0.y = pack2(acc0[2]*rl, acc0[3]*rl);
  o0.z = pack2(acc0[4]*rl, acc0[5]*rl); o0.w = pack2(acc0[6]*rl, acc0[7]*rl);
  o1.x = pack2(acc1[0]*rl, acc1[1]*rl); o1.y = pack2(acc1[2]*rl, acc1[3]*rl);
  o1.z = pack2(acc1[4]*rl, acc1[5]*rl); o1.w = pack2(acc1[6]*rl, acc1[7]*rl);
  unsigned short* orow = O + (size_t)q * EMB + hb;
  *reinterpret_cast<uint4*>(orow + 8 * lh) = o0;          // d = 8*lh .. +7
  *reinterpret_cast<uint4*>(orow + 16 + 8 * lh) = o1;     // d = 16+8*lh .. +7
}

// ---------------------------------------------------------------------------
extern "C" void kernel_launch(void* const* d_in, const int* in_sizes, int n_in,
                              void* d_out, int out_size, void* d_ws, size_t ws_size,
                              hipStream_t stream) {
  const float* x        = (const float*)d_in[0];
  const float* context  = (const float*)d_in[1];
  const unsigned char* pad_mask = (const unsigned char*)d_in[2];  // bool(1B)
  const float* row_embed = (const float*)d_in[3];
  const float* col_embed = (const float*)d_in[4];
  // d_in[5]=w_in, [6]=b_in
  const float* w_in  = (const float*)d_in[5];
  const float* b_in  = (const float*)d_in[6];
  const float* wq    = (const float*)d_in[7];
  const float* bq    = (const float*)d_in[8];
  const float* wk    = (const float*)d_in[9];
  const float* bk    = (const float*)d_in[10];
  const float* wv    = (const float*)d_in[11];
  const float* bv    = (const float*)d_in[12];
  const float* w_out = (const float*)d_in[13];
  const float* b_out = (const float*)d_in[14];
  float* out = (float*)d_out;

  char* ws = (char*)d_ws;
  const size_t TOK = (size_t)NPAD * EMB * sizeof(unsigned short); // 2,572,288
  const size_t WSZ = (size_t)EMB * EMB * sizeof(unsigned short);  // 131,072
  size_t off = 0;
  unsigned short* A_x = (unsigned short*)(ws + off); off += TOK;
  unsigned short* A_c = (unsigned short*)(ws + off); off += TOK;
  unsigned short* Qbf = (unsigned short*)(ws + off); off += TOK;
  unsigned short* Kbf = (unsigned short*)(ws + off); off += TOK;
  unsigned short* Vt  = (unsigned short*)(ws + off); off += TOK;
  unsigned short* Wq  = (unsigned short*)(ws + off); off += WSZ;
  unsigned short* Wk  = (unsigned short*)(ws + off); off += WSZ;
  unsigned short* Wv  = (unsigned short*)(ws + off); off += WSZ;
  unsigned short* Wo  = (unsigned short*)(ws + off); off += WSZ;
  float* bq2 = (float*)(ws + off); off += 1024;
  float* bk2 = (float*)(ws + off); off += 1024;
  float* bv2 = (float*)(ws + off); off += 1024;
  unsigned short* O = A_x;   // A_x is dead after Q GEMM; reuse for attn output

  prep_weights<<<dim3(EMB), dim3(256), 0, stream>>>(
      wq, bq, wk, bk, wv, bv, w_in, b_in, w_out, Wq, Wk, Wv, Wo, bq2, bk2, bv2);
  prep_tokens<<<dim3(NPAD / 32, EMB / 32, 2), dim3(256), 0, stream>>>(
      x, context, row_embed, col_embed, A_x, A_c);
  gemm_ne<<<dim3(NPAD / 16), dim3(256), 0, stream>>>(A_x, Wq, bq2, 0.0625f, Qbf); // scale EMB^-0.5
  gemm_ne<<<dim3(NPAD / 16), dim3(256), 0, stream>>>(A_c, Wk, bk2, 1.0f, Kbf);
  gemm_vt<<<dim3(NPAD / 16), dim3(256), 0, stream>>>(Wv, A_c, bv2, Vt);
  flash_attn<<<dim3((NTOK + 15) / 16), dim3(256), 0, stream>>>(Qbf, Kbf, Vt, pad_mask, O);
  gemm_out<<<dim3((NTOK + 15) / 16), dim3(256), 0, stream>>>(Wo, O, b_out, out);
}
